// LightPAM_45191645888896
// MI455X (gfx1250) — compile-verified, matrix operand order checked by reading, not run on hardware
//
#include <hip/hip_runtime.h>
#include <hip/hip_bf16.h>

// ---------------------------------------------------------------------------
// LightPAM for MI455X (gfx1250): wave32 + v_wmma_f32_16x16x32_f16 for every
// GEMM shape (1x1 conv, Q.K score, M.x warp); async global->LDS staging for
// the grouped 3x3 convs; 320KB-LDS mega-kernel holds the whole 256x256
// attention matrix on-chip per (b,h) row. Q/K stored pixel-major (c contig)
// so WMMA fragment gathers are contiguous 16B runs instead of 256KB strides.
// ---------------------------------------------------------------------------

#define B_  2
#define C_  64
#define H_  256
#define W_  256
#define G_  4
#define CG_ 16               // C_/G_
#define HW_ (H_*W_)          // 65536
#define NE_ ((long)B_*C_*H_*W_)   // 8,388,608 elements per tensor

typedef __attribute__((ext_vector_type(16))) _Float16 v16h;
typedef __attribute__((ext_vector_type(8)))  float    v8f;

#if __has_builtin(__builtin_amdgcn_global_load_async_to_lds_b128)
#define HAVE_ASYNC_LDS 1
typedef int v4i_ __attribute__((__vector_size__(4 * sizeof(int))));
typedef __attribute__((address_space(1))) v4i_* g_v4i_ptr;
typedef __attribute__((address_space(3))) v4i_* l_v4i_ptr;
#else
#define HAVE_ASYNC_LDS 0
#endif

__device__ __forceinline__ void async_wait_all() {
#if HAVE_ASYNC_LDS
#if __has_builtin(__builtin_amdgcn_s_wait_asynccnt)
    __builtin_amdgcn_s_wait_asynccnt(0);
#else
    asm volatile("s_wait_asynccnt 0x0" ::: "memory");
#endif
#endif
}

// copy 16B global -> LDS (async on gfx1250 toolchains that expose it)
__device__ __forceinline__ void copy16_g2l(const float* g, float* l) {
#if HAVE_ASYNC_LDS
    __builtin_amdgcn_global_load_async_to_lds_b128(
        (g_v4i_ptr)(uintptr_t)g, (l_v4i_ptr)(uintptr_t)l, 0, 0);
#else
    *(float4*)l = *(const float4*)g;
#endif
}

// A/B fragment K index for 16-bit WMMA 16x16x32 (ISA 7.12.2):
// lane 0-15 / 16-31 split; element e in lane's 16 halfs.
__device__ __forceinline__ int kmap(int e, int kg) {
    return ((e < 8) ? e : (e + 8)) + (kg << 3);
}

// ---------------------------------------------------------------------------
// Kernel 1: BN + grouped 3x3 conv + leaky_relu(0.1)  -> h   (NCHW)
// One block = one (b,o,y) row. 16ch x 3rows x 256px staged in LDS (48KB).
// ---------------------------------------------------------------------------
__global__ __launch_bounds__(256)
void k_bnconv1(const float* __restrict__ x,
               const float* __restrict__ gamma, const float* __restrict__ beta,
               const float* __restrict__ mean,  const float* __restrict__ var,
               const float* __restrict__ w1,    const float* __restrict__ b1,
               float* __restrict__ hout)
{
    __shared__ __align__(16) float inbuf[CG_*3*W_];
    __shared__ float wsm[CG_*9];
    __shared__ float invs[CG_], adds[CG_];
    const int tid = threadIdx.x;
    const long idx = (long)blockIdx.x * 256 + tid;
    const int b  = (int)(idx >> 22);          // / (C*H*W)
    const int o  = (int)((idx >> 16) & 63);   // channel (uniform per block)
    const int y  = (int)((idx >> 8) & 255);   // row (uniform per block)
    const int xw = (int)(idx & 255);
    const int cb = (o >> 4) << 4;             // group base channel

    if (tid < CG_*9) wsm[tid] = w1[o*CG_*9 + tid];
    if (tid < CG_) {
        int c = cb + tid;
        float inv = gamma[c] * rsqrtf(var[c] + 1e-5f);
        invs[tid] = inv;
        adds[tid] = beta[c] - mean[c] * inv;
    }

    // stage the 16-channel, 3-row input stripe
    const float* xb = x + ((long)(b*C_) << 16);
    for (int q = tid; q < CG_*3*(W_/4); q += 256) {   // float4 granules
        int ci  = q / (3*(W_/4));
        int rem = q % (3*(W_/4));
        int row = rem / (W_/4);
        int seg = rem % (W_/4);
        float* l = &inbuf[(ci*3 + row)*W_ + seg*4];
        int yy = y + row - 1;
        if ((unsigned)yy < (unsigned)H_) {
            copy16_g2l(xb + ((long)(cb + ci) << 16) + (long)yy*W_ + seg*4, l);
        } else {
            *(float4*)l = make_float4(0.f, 0.f, 0.f, 0.f);
        }
    }
    async_wait_all();
    __syncthreads();

    float acc = b1[o];
    for (int ci = 0; ci < CG_; ++ci) {
        const float inv = invs[ci], add = adds[ci];
        const float* lrow = &inbuf[ci*3*W_];
        #pragma unroll
        for (int ky = 0; ky < 3; ++ky) {
            if ((unsigned)(y + ky - 1) >= (unsigned)H_) continue;  // zero-pad row
            #pragma unroll
            for (int kx = 0; kx < 3; ++kx) {
                int xx = xw + kx - 1;
                if ((unsigned)xx >= (unsigned)W_) continue;        // zero-pad col
                acc += wsm[ci*9 + ky*3 + kx] * (lrow[ky*W_ + xx] * inv + add);
            }
        }
    }
    hout[idx] = acc > 0.f ? acc : 0.1f * acc;
}

// ---------------------------------------------------------------------------
// Kernel 2: grouped 3x3 conv on h + residual BN(x) -> r, emitted CHANNEL-LAST
// (b,p,c) so the 1x1 WMMA B-fragments read contiguous floats.
// ---------------------------------------------------------------------------
__global__ __launch_bounds__(256)
void k_conv2res(const float* __restrict__ hin, const float* __restrict__ x,
                const float* __restrict__ gamma, const float* __restrict__ beta,
                const float* __restrict__ mean,  const float* __restrict__ var,
                const float* __restrict__ w2,    const float* __restrict__ b2,
                float* __restrict__ rout)
{
    __shared__ __align__(16) float inbuf[CG_*3*W_];
    __shared__ float wsm[CG_*9];
    const int tid = threadIdx.x;
    const long idx = (long)blockIdx.x * 256 + tid;
    const int b  = (int)(idx >> 22);
    const int o  = (int)((idx >> 16) & 63);
    const int y  = (int)((idx >> 8) & 255);
    const int xw = (int)(idx & 255);
    const int cb = (o >> 4) << 4;

    if (tid < CG_*9) wsm[tid] = w2[o*CG_*9 + tid];

    const float* hb = hin + ((long)(b*C_) << 16);
    for (int q = tid; q < CG_*3*(W_/4); q += 256) {
        int ci  = q / (3*(W_/4));
        int rem = q % (3*(W_/4));
        int row = rem / (W_/4);
        int seg = rem % (W_/4);
        float* l = &inbuf[(ci*3 + row)*W_ + seg*4];
        int yy = y + row - 1;
        if ((unsigned)yy < (unsigned)H_) {
            copy16_g2l(hb + ((long)(cb + ci) << 16) + (long)yy*W_ + seg*4, l);
        } else {
            *(float4*)l = make_float4(0.f, 0.f, 0.f, 0.f);
        }
    }
    async_wait_all();
    __syncthreads();

    float acc = b2[o];
    for (int ci = 0; ci < CG_; ++ci) {
        const float* lrow = &inbuf[ci*3*W_];
        #pragma unroll
        for (int ky = 0; ky < 3; ++ky) {
            if ((unsigned)(y + ky - 1) >= (unsigned)H_) continue;
            #pragma unroll
            for (int kx = 0; kx < 3; ++kx) {
                int xx = xw + kx - 1;
                if ((unsigned)xx >= (unsigned)W_) continue;
                acc += wsm[ci*9 + ky*3 + kx] * lrow[ky*W_ + xx];
            }
        }
    }
    float inv = gamma[o] * rsqrtf(var[o] + 1e-5f);
    float bnx = x[idx] * inv + (beta[o] - mean[o] * inv);
    // channel-last store: (b, p=y*W+xw, c=o)
    rout[((long)b*HW_ + y*W_ + xw)*C_ + o] = acc + bnx;
}

// ---------------------------------------------------------------------------
// Kernel 3: 1x1 conv as WMMA GEMM: Y[b,o,p] = sum_c Wt[o,c]*R[b,p,c] + bias.
// R channel-last -> B fragments are contiguous. Y emitted NCHW for meansub.
// ---------------------------------------------------------------------------
__global__ __launch_bounds__(256)
void k_conv1x1_wmma(const float* __restrict__ R, const float* __restrict__ Wt,
                    const float* __restrict__ bias, float* __restrict__ Y)
{
    const int wave = threadIdx.x >> 5;
    const int lane = threadIdx.x & 31;
    const int ln = lane & 15, kg = lane >> 4;
    const int tile = blockIdx.x * 8 + wave;       // B*(C/16)*(HW/16) = 32768 tiles
    const int pt = tile & 4095;                   // pixel tile
    const int ot = (tile >> 12) & 3;              // out-channel tile
    const int b  = tile >> 14;
    const int p  = pt * 16 + ln;
    const float* Rb = R + (long)b * HW_ * C_;

    v8f acc = {};
    #pragma unroll
    for (int ch = 0; ch < 2; ++ch) {
        v16h a, bf;
        #pragma unroll
        for (int e = 0; e < 16; ++e) {
            int c = ch * 32 + kmap(e, kg);
            a[e]  = (_Float16)Wt[(ot*16 + ln) * C_ + c];   // A: rows o, K=c
            bf[e] = (_Float16)Rb[(long)p * C_ + c];        // B: K=c, cols p (contig)
        }
        acc = __builtin_amdgcn_wmma_f32_16x16x32_f16(false, a, false, bf,
                                                     (short)0, acc, false, false);
    }
    #pragma unroll
    for (int v = 0; v < 8; ++v) {
        int o = ot*16 + v + (kg << 3);
        Y[((long)(b*C_ + o)) * HW_ + p] = acc[v] + bias[o];
    }
}

// ---------------------------------------------------------------------------
// Kernel 4: subtract per-(b,c,h) row mean over W; emit fp16 TRANSPOSED to
// pixel-major (b,h,i,c) so score-phase WMMA fragments load contiguous halfs.
// One wave per row.
// ---------------------------------------------------------------------------
__global__ __launch_bounds__(256)
void k_meansub_f16t(const float* __restrict__ Yf, _Float16* __restrict__ Yt)
{
    const int row  = blockIdx.x * 8 + (threadIdx.x >> 5);  // (b*C+c)*H + hr
    const int lane = threadIdx.x & 31;
    const int b  = row >> 14;
    const int c  = (row >> 8) & 63;
    const int hr = row & 255;
    const float* src = Yf + (long)row * W_;
    float s = 0.f;
    #pragma unroll
    for (int i = 0; i < W_/32; ++i) s += src[lane + 32*i];
    #pragma unroll
    for (int m = 16; m >= 1; m >>= 1) s += __shfl_xor(s, m, 32);
    const float mean = s * (1.0f / W_);
    _Float16* dst = Yt + ((long)(b*H_ + hr) * W_) * C_ + c;
    #pragma unroll
    for (int i = 0; i < W_/32; ++i) {
        int px = lane + 32*i;
        dst[(long)px * C_] = (_Float16)(src[px] - mean);
    }
}

// ---------------------------------------------------------------------------
// Kernel 5: attention mega-kernel. One block per n=(b,h). Full 256x256 score
// matrix lives in LDS (272KB of the 320KB WGP budget). Phases:
//   A) S = Q.K^T via WMMA (Q/K pixel-major fp16)
//   B) row/col softmax stats; S := M_r2l in place
//   C) V_left/V_right (relaxed dual-softmax consistency) + tanh gate
//   D) x_leftT = M_r2l.x_right, x_rightT = M_l2r.x_left via WMMA; blend.
// M_l2r recovered: M_l2r[j,i] = S[i,j]*rsum[i]*exp(rmax[i]-cmax[j])/csum[j]
// ---------------------------------------------------------------------------
#define SP_ 260   // padded LDS row stride (floats)
#define ATTN_SMEM_BYTES ((256*SP_ + 6*256) * sizeof(float))

extern __shared__ float smem[];

__global__ __launch_bounds__(256)
void k_pam_attn(const _Float16* __restrict__ Qt, const _Float16* __restrict__ Kt,
                const float* __restrict__ xl, const float* __restrict__ xr,
                float* __restrict__ out)
{
    float* S    = smem;
    float* rmax = smem + 256*SP_;
    float* rsum = rmax + 256;
    float* cmax = rsum + 256;
    float* csum = cmax + 256;
    float* vlb  = csum + 256;   // tanh(5*V_left)
    float* vrb  = vlb  + 256;   // tanh(5*V_right)

    const int n  = blockIdx.x;
    const int b  = n >> 8;
    const int hr = n & 255;
    const int tid  = threadIdx.x;
    const int wave = tid >> 5;
    const int lane = tid & 31;
    const int ln = lane & 15, kg = lane >> 4;

    const _Float16* qb = Qt + ((long)(b*H_ + hr) * W_) * C_;  // (i,c) -> qb[i*C+c]
    const _Float16* kb = Kt + ((long)(b*H_ + hr) * W_) * C_;
    const float*   xlb = xl + (long)b*C_*HW_ + (long)hr*W_;   // (c,i) -> xlb[c*HW+i]
    const float*   xrb = xr + (long)b*C_*HW_ + (long)hr*W_;

    // ---- Phase A: score S[i,j] = sum_c Q[i,c]*K[j,c] -------------------
    for (int t = wave; t < 256; t += 8) {
        const int ti = t >> 4, tj = t & 15;
        v8f acc = {};
        #pragma unroll
        for (int ch = 0; ch < 2; ++ch) {
            v16h a, bf;
            #pragma unroll
            for (int e = 0; e < 16; ++e) {
                int c = ch * 32 + kmap(e, kg);
                a[e]  = qb[(long)(ti*16 + ln) * C_ + c];   // contiguous runs
                bf[e] = kb[(long)(tj*16 + ln) * C_ + c];
            }
            acc = __builtin_amdgcn_wmma_f32_16x16x32_f16(false, a, false, bf,
                                                         (short)0, acc, false, false);
        }
        #pragma unroll
        for (int v = 0; v < 8; ++v)
            S[(ti*16 + v + (kg<<3))*SP_ + tj*16 + ln] = acc[v];
    }
    __syncthreads();

    // ---- Phase B: softmax stats, then S := M_r2l in place --------------
    {
        const int i = tid;
        float m = -3.4e38f;
        for (int j = 0; j < 256; ++j) m = fmaxf(m, S[i*SP_ + j]);
        float s = 0.f;
        for (int j = 0; j < 256; ++j) s += __expf(S[i*SP_ + j] - m);
        rmax[i] = m; rsum[i] = s;
        m = -3.4e38f;
        for (int j = 0; j < 256; ++j) m = fmaxf(m, S[j*SP_ + i]);
        s = 0.f;
        for (int j = 0; j < 256; ++j) s += __expf(S[j*SP_ + i] - m);
        cmax[i] = m; csum[i] = s;
    }
    __syncthreads();
    {
        const int i = tid;
        const float m = rmax[i], rs = 1.f / rsum[i];
        for (int j = 0; j < 256; ++j)
            S[i*SP_ + j] = __expf(S[i*SP_ + j] - m) * rs;
    }
    __syncthreads();

    // ---- Phase C: V_left / V_right (relaxed masks) ---------------------
    {
        const int i = tid;
        const float rm_i = rmax[i], rs_i = rsum[i], cm_i = cmax[i];
        // V_left[i] = sum_j (sum_d M_r2l[i+d,j]) * M_l2r[j,i]
        float vl = 0.f;
        for (int j = 0; j < 256; ++j) {
            float rel = 0.f;
            #pragma unroll
            for (int d = -2; d <= 2; ++d) {
                int q = i + d;
                if ((unsigned)q < 256u) rel += S[q*SP_ + j];
            }
            float ml2r = S[i*SP_ + j] * rs_i * __expf(rm_i - cmax[j]) / csum[j];
            vl += rel * ml2r;
        }
        vlb[i] = tanhf(5.f * vl);
        // V_right[i] = sum_j (sum_d M_l2r[i+d,j]) * M_r2l[j,i]
        float einv[5];
        #pragma unroll
        for (int d = 0; d < 5; ++d) {
            int q = i + d - 2;
            einv[d] = ((unsigned)q < 256u) ? __expf(cm_i - cmax[q]) / csum[q] : 0.f;
        }
        float vr = 0.f;
        for (int j = 0; j < 256; ++j) {
            float rel = 0.f;
            #pragma unroll
            for (int d = 0; d < 5; ++d) {
                int q = i + d - 2;
                if ((unsigned)q < 256u) rel += S[j*SP_ + q] * einv[d];
            }
            vr += (rsum[j] * __expf(rmax[j] - cm_i)) * rel * S[j*SP_ + i];
        }
        vrb[i] = tanhf(5.f * vr);
    }
    __syncthreads();

    // ---- Phase D: warp GEMMs + blend -----------------------------------
    const long CHWB = NE_;
    // x_leftT = M_r2l . x_right ; out_left blend
    for (int t = wave; t < 64; t += 8) {
        const int ti = t >> 2, tc = t & 3;
        v8f acc = {};
        #pragma unroll
        for (int jc = 0; jc < 8; ++jc) {
            v16h a, bf;
            #pragma unroll
            for (int e = 0; e < 16; ++e) {
                int j = jc * 32 + kmap(e, kg);
                a[e]  = (_Float16)S[(ti*16 + ln)*SP_ + j];          // M_r2l[i,j]
                bf[e] = (_Float16)xrb[(long)(tc*16 + ln)*HW_ + j];  // x_right[j,c]
            }
            acc = __builtin_amdgcn_wmma_f32_16x16x32_f16(false, a, false, bf,
                                                         (short)0, acc, false, false);
        }
        #pragma unroll
        for (int v = 0; v < 8; ++v) {
            int i = ti*16 + v + (kg<<3);
            int c = tc*16 + ln;
            float vt = vlb[i];
            long off = ((long)(b*C_ + c))*HW_ + (long)hr*W_ + i;
            out[off] = xlb[(long)c*HW_ + i] * (1.f - vt) + acc[v] * vt;
        }
    }
    // x_rightT = M_l2r . x_left ; out_right blend
    for (int t = wave; t < 64; t += 8) {
        const int ti = t >> 2, tc = t & 3;
        const int r  = ti*16 + ln;
        const float cmr = cmax[r], csr = 1.f / csum[r];
        v8f acc = {};
        #pragma unroll
        for (int jc = 0; jc < 8; ++jc) {
            v16h a, bf;
            #pragma unroll
            for (int e = 0; e < 16; ++e) {
                int j = jc * 32 + kmap(e, kg);
                // M_l2r[r,j] = S[j,r]*rsum[j]*exp(rmax[j]-cmax[r])/csum[r]
                a[e]  = (_Float16)(S[j*SP_ + r] * rsum[j] * __expf(rmax[j] - cmr) * csr);
                bf[e] = (_Float16)xlb[(long)(tc*16 + ln)*HW_ + j];  // x_left[j,c]
            }
            acc = __builtin_amdgcn_wmma_f32_16x16x32_f16(false, a, false, bf,
                                                         (short)0, acc, false, false);
        }
        #pragma unroll
        for (int v = 0; v < 8; ++v) {
            int i = ti*16 + v + (kg<<3);
            int c = tc*16 + ln;
            float vt = vrb[i];
            long off = CHWB + ((long)(b*C_ + c))*HW_ + (long)hr*W_ + i;
            out[off] = xrb[(long)c*HW_ + i] * (1.f - vt) + acc[v] * vt;
        }
    }
}

// ---------------------------------------------------------------------------
// Host launcher
// ---------------------------------------------------------------------------
extern "C" void kernel_launch(void* const* d_in, const int* in_sizes, int n_in,
                              void* d_out, int out_size, void* d_ws, size_t ws_size,
                              hipStream_t stream)
{
    (void)in_sizes; (void)n_in; (void)out_size; (void)ws_size;
    const float* x_left   = (const float*)d_in[0];
    const float* x_right  = (const float*)d_in[1];
    // d_in[2], d_in[3] (catfea_*) unused by reference math
    const float* bn_gamma = (const float*)d_in[4];
    const float* bn_beta  = (const float*)d_in[5];
    const float* bn_mean  = (const float*)d_in[6];
    const float* bn_var   = (const float*)d_in[7];
    const float* rb_w1    = (const float*)d_in[8];
    const float* rb_b1    = (const float*)d_in[9];
    const float* rb_w2    = (const float*)d_in[10];
    const float* rb_b2    = (const float*)d_in[11];
    const float* bq_w     = (const float*)d_in[12];
    const float* bq_b     = (const float*)d_in[13];
    const float* bs_w     = (const float*)d_in[14];
    const float* bs_b     = (const float*)d_in[15];
    float* out = (float*)d_out;

    // Workspace (floats): [tmp | r_left | r_right | Qt(f16) Kt(f16)]
    float* ws   = (float*)d_ws;
    float* tmp  = ws;                 // h scratch, then Qf, then Kf (fp32)
    float* rL   = ws + NE_;           // channel-last (b,p,c)
    float* rR   = ws + 2*NE_;
    _Float16* Qt = (_Float16*)(ws + 3*NE_);   // pixel-major (b,h,i,c) fp16
    _Float16* Kt = Qt + NE_;

    const int convBlocks = (int)(NE_ / 256);   // 32768

    // Left branch: BN+conv1+leaky -> tmp ; conv2+residual -> rL (chan-last)
    k_bnconv1<<<convBlocks, 256, 0, stream>>>(x_left, bn_gamma, bn_beta, bn_mean,
                                              bn_var, rb_w1, rb_b1, tmp);
    k_conv2res<<<convBlocks, 256, 0, stream>>>(tmp, x_left, bn_gamma, bn_beta,
                                               bn_mean, bn_var, rb_w2, rb_b2, rL);
    // Right branch
    k_bnconv1<<<convBlocks, 256, 0, stream>>>(x_right, bn_gamma, bn_beta, bn_mean,
                                              bn_var, rb_w1, rb_b1, tmp);
    k_conv2res<<<convBlocks, 256, 0, stream>>>(tmp, x_right, bn_gamma, bn_beta,
                                               bn_mean, bn_var, rb_w2, rb_b2, rR);

    // Q = bq(rL); mean-subtract -> fp16 transposed
    k_conv1x1_wmma<<<4096, 256, 0, stream>>>(rL, bq_w, bq_b, tmp);
    k_meansub_f16t<<<4096, 256, 0, stream>>>(tmp, Qt);
    // K = bs(rR); mean-subtract -> fp16 transposed
    k_conv1x1_wmma<<<4096, 256, 0, stream>>>(rR, bs_w, bs_b, tmp);
    k_meansub_f16t<<<4096, 256, 0, stream>>>(tmp, Kt);

    // Attention mega-kernel: 272KB dynamic LDS per block (<= 320KB WGP budget)
    (void)hipFuncSetAttribute((const void*)k_pam_attn,
                              hipFuncAttributeMaxDynamicSharedMemorySize,
                              (int)ATTN_SMEM_BYTES);
    k_pam_attn<<<B_*H_, 256, ATTN_SMEM_BYTES, stream>>>(Qt, Kt, x_left, x_right, out);
}
